// DynamicFusionModule_27127013441759
// MI455X (gfx1250) — matrix operand order repaired and checked by reading, not compiled, not run on hardware
//
#include <hip/hip_runtime.h>

// ---------------------------------------------------------------------------
// Types and WMMA helpers (CDNA5 / gfx1250, wave32)
// ---------------------------------------------------------------------------
typedef _Float16 v16h __attribute__((ext_vector_type(16)));
typedef _Float16 v8h  __attribute__((ext_vector_type(8)));
typedef float    v8f  __attribute__((ext_vector_type(8)));
typedef int      v4i  __attribute__((vector_size(16)));   // matches async-LDS builtin param

#define WMMA_F32_F16(a, b, c) \
  __builtin_amdgcn_wmma_f32_16x16x32_f16(false, (a), false, (b), (short)0, (c), false, false)

// Async global->LDS (CDNA5, ASYNCcnt-tracked); guarded, with sync fallback.
#if defined(__has_builtin)
#if __has_builtin(__builtin_amdgcn_global_load_async_to_lds_b128) && \
    __has_builtin(__builtin_amdgcn_s_wait_asynccnt)
#define HAVE_ASYNC_LDS 1
#endif
#endif
#ifndef HAVE_ASYNC_LDS
#define HAVE_ASYNC_LDS 0
#endif

// A-fragment: 16x32 f16, row-major source. Lane L holds row m=L&15;
// K pairs per ISA table: lanes 0-15 K={0..7,16..23}, lanes 16-31 K={8..15,24..31}.
static __device__ __forceinline__ v16h load_a_frag(const _Float16* __restrict__ A, int ld) {
  const int lane = threadIdx.x & 31;
  const int m = lane & 15;
  const int hg = lane >> 4;
  v16h a;
#pragma unroll
  for (int g = 0; g < 8; ++g) {
    const int kb = ((g & 3) << 1) + hg * 8 + ((g >> 2) << 4);
    const _Float16* p = A + (size_t)m * ld + kb;
    a[2 * g]     = p[0];
    a[2 * g + 1] = p[1];
  }
  return a;
}

// B-fragment: 32x16 f16. Lane L holds column n=L&15; K per ISA table (same as A).
static __device__ __forceinline__ v16h load_b_frag(const _Float16* __restrict__ Bt, int ldb) {
  const int lane = threadIdx.x & 31;
  const int hg = lane >> 4;
  const _Float16* p = Bt + (lane & 15);
  v16h bf;
#pragma unroll
  for (int s = 0; s < 16; ++s) {
    const int k = (s & 7) + hg * 8 + ((s >> 3) << 4);
    bf[s] = p[(size_t)k * ldb];
  }
  return bf;
}

// C/D layout: element (r, n): m = r + 8*(lane>>4), n = lane&15.

// ---------------------------------------------------------------------------
// Constants
// ---------------------------------------------------------------------------
#define NB   2          // batch
#define CCH  128        // channels
#define C2   64         // agent hidden
#define NPIX 65536      // H*W
#define NTOK 4096       // K tokens
#define EPSV 1e-5f

// ---------------------------------------------------------------------------
// Small init kernel: zero histogram / counters / aux_loss
// ---------------------------------------------------------------------------
__global__ void zero_kernel(int* hist, int* counters, float* aux) {
  const int i = blockIdx.x * 256 + threadIdx.x;
  if (i < NB * 4096) hist[i] = 0;
  if (i < 8) counters[i] = 0;
  if (i == 0) *aux = 0.0f;
}

// ---------------------------------------------------------------------------
// Weight prep: transpose [N][K] f32 -> [K][N] f16 (B-matrix layout)
// ---------------------------------------------------------------------------
struct PrepJobs {
  const float* src[9];
  _Float16*    dst[9];
  int          rows[9];   // N (output features)
  int          cols[9];   // K (input features)
};

__global__ void prep_kernel(PrepJobs jobs) {
  const int j = blockIdx.y;
  const float* src = jobs.src[j];
  _Float16* dst = jobs.dst[j];
  const int rows = jobs.rows[j], cols = jobs.cols[j];
  for (int i = blockIdx.x * 256 + threadIdx.x; i < rows * cols; i += gridDim.x * 256) {
    const int n = i / cols, k = i % cols;
    dst[(size_t)k * rows + n] = (_Float16)src[i];
  }
}

// ---------------------------------------------------------------------------
// Base output: out[b][c][n] = f_ir + f_vis   (16.7M f32, HBM-bound)
// ---------------------------------------------------------------------------
__global__ void basesum_kernel(const float* __restrict__ a, const float* __restrict__ b,
                               float* __restrict__ o) {
  const size_t i = ((size_t)blockIdx.x * 256 + threadIdx.x) * 4;
  const float4 x = *(const float4*)(a + i);
  const float4 y = *(const float4*)(b + i);
  float4 r; r.x = x.x + y.x; r.y = x.y + y.y; r.z = x.z + y.z; r.w = x.w + y.w;
  *(float4*)(o + i) = r;
}

// ---------------------------------------------------------------------------
// Agent stage 1: 1x1 conv (256 -> 64) + BN1 + ReLU, via WMMA.
// 64-pixel tile per 128-thread block (33.8KB LDS, safely under 64KB).
// ---------------------------------------------------------------------------
__global__ void __launch_bounds__(128) agent1_kernel(
    const float* __restrict__ fir, const float* __restrict__ fvis,
    const _Float16* __restrict__ w1t,  // [256][64] f16
    const float* __restrict__ bnb, const float* __restrict__ bng,
    const float* __restrict__ bnm, const float* __restrict__ bnv,
    float* __restrict__ x1) {          // [B][64][NPIX]
  extern __shared__ char smem_raw[];
  _Float16* lds = (_Float16*)smem_raw;        // [64][264]
  const int ld = 264;
  const int b = blockIdx.y;
  const size_t pbase = (size_t)blockIdx.x * 64;

  for (int i = threadIdx.x; i < 64 * 256; i += 128) {
    const int c = i >> 6;        // channel 0..255
    const int m = i & 63;        // pixel within tile
    const float val = (c < CCH)
        ? fir[((size_t)b * CCH + c) * NPIX + pbase + m]
        : fvis[((size_t)b * CCH + (c - CCH)) * NPIX + pbase + m];
    lds[(size_t)m * ld + c] = (_Float16)val;
  }
  __syncthreads();

  const int wave = threadIdx.x >> 5;
  const int lane = threadIdx.x & 31;
  const _Float16* Al = lds + (size_t)wave * 16 * ld;

  for (int nt = 0; nt < 4; ++nt) {
    v8f acc = {0, 0, 0, 0, 0, 0, 0, 0};
    for (int kc = 0; kc < 8; ++kc) {
      v16h a = load_a_frag(Al + kc * 32, ld);
      v16h bf = load_b_frag(w1t + (size_t)(kc * 32) * 64 + nt * 16, 64);
      acc = WMMA_F32_F16(a, bf, acc);
    }
    const int n = nt * 16 + (lane & 15);
    const float sc = bng[n] * rsqrtf(bnv[n] + EPSV);
    const float sh = bnb[n] - bnm[n] * sc;
    float* dst = x1 + ((size_t)b * C2 + n) * NPIX + pbase + wave * 16 + ((lane >> 4) << 3);
#pragma unroll
    for (int r = 0; r < 8; ++r) dst[r] = fmaxf(acc[r] * sc + sh, 0.0f);
  }
}

// ---------------------------------------------------------------------------
// Agent stage 2: depthwise 3x3 + BN2 + ReLU + (64->1) proj + sigmoid
// ---------------------------------------------------------------------------
__global__ void __launch_bounds__(256) agent2_kernel(
    const float* __restrict__ x1, const float* __restrict__ wdw,
    const float* __restrict__ bnb, const float* __restrict__ bng,
    const float* __restrict__ bnm, const float* __restrict__ bnv,
    const float* __restrict__ w3, const float* __restrict__ b3,
    float* __restrict__ wmap, float* __restrict__ sal) {
  __shared__ float t[18][19];
  const int tx = threadIdx.x & 15, ty = threadIdx.x >> 4;
  const int gx = blockIdx.x * 16, gy = blockIdx.y * 16, b = blockIdx.z;
  float accum = 0.0f;
  for (int c = 0; c < C2; ++c) {
    const float* src = x1 + ((size_t)b * C2 + c) * NPIX;
    for (int i = threadIdx.x; i < 18 * 18; i += 256) {
      const int yy = i / 18 - 1 + gy, xx = i % 18 - 1 + gx;
      t[i / 18][i % 18] = (yy >= 0 && yy < 256 && xx >= 0 && xx < 256) ? src[yy * 256 + xx] : 0.0f;
    }
    __syncthreads();
    float s = 0.0f;
#pragma unroll
    for (int ky = 0; ky < 3; ++ky)
#pragma unroll
      for (int kx = 0; kx < 3; ++kx) s += t[ty + ky][tx + kx] * wdw[c * 9 + ky * 3 + kx];
    const float sc = bng[c] * rsqrtf(bnv[c] + EPSV);
    s = fmaxf((s - bnm[c]) * sc + bnb[c], 0.0f);
    accum += w3[c] * s;
    __syncthreads();
  }
  const float wv = 1.0f / (1.0f + __expf(-(accum + b3[0])));
  const int n = (gy + ty) * 256 + gx + tx;
  wmap[(size_t)b * NPIX + n] = wv;
  sal[(size_t)b * NPIX + n] = fabsf(wv - 0.5f);
}

// ---------------------------------------------------------------------------
// Exact top-K via histogram select (sal in [0,0.5))
// ---------------------------------------------------------------------------
__global__ void hist_kernel(const float* __restrict__ sal, int* __restrict__ hist) {
  const int b = blockIdx.y;
  const int n = blockIdx.x * 256 + threadIdx.x;
  int bin = (int)(sal[(size_t)b * NPIX + n] * 8192.0f);
  bin = bin > 4095 ? 4095 : (bin < 0 ? 0 : bin);
  atomicAdd(&hist[b * 4096 + bin], 1);
}

__global__ void thresh_kernel(const int* __restrict__ hist, int* __restrict__ tinfo) {
  if (threadIdx.x != 0) return;
  const int b = blockIdx.x;
  int cum = 0, bin = 0;
  for (int i = 4095; i >= 0; --i) {
    const int h = hist[b * 4096 + i];
    if (cum + h >= NTOK) { bin = i; break; }
    cum += h;
  }
  tinfo[b * 4 + 0] = bin;        // threshold bin
  tinfo[b * 4 + 1] = cum;        // count strictly above
  tinfo[b * 4 + 2] = NTOK - cum; // take from threshold bin
}

__global__ void compact_kernel(const float* __restrict__ sal, const int* __restrict__ tinfo,
                               int* __restrict__ counters, int* __restrict__ idx) {
  const int b = blockIdx.y;
  const int n = blockIdx.x * 256 + threadIdx.x;
  int bin = (int)(sal[(size_t)b * NPIX + n] * 8192.0f);
  bin = bin > 4095 ? 4095 : (bin < 0 ? 0 : bin);
  const int tb = tinfo[b * 4];
  if (bin > tb) {
    const int p = atomicAdd(&counters[b * 2], 1);
    idx[b * NTOK + p] = n;
  } else if (bin == tb) {
    const int p = atomicAdd(&counters[b * 2 + 1], 1);
    if (p < tinfo[b * 4 + 2]) idx[b * NTOK + tinfo[b * 4 + 1] + p] = n;
  }
}

// ---------------------------------------------------------------------------
// Gather selected tokens: X[sb][tok][c] (sb = stream*2 + b; stream0=ir)
// ---------------------------------------------------------------------------
__global__ void gather_kernel(const float* __restrict__ fir, const float* __restrict__ fvis,
                              const float* __restrict__ wmap, const int* __restrict__ idx,
                              float* __restrict__ X, float* __restrict__ selw) {
  const int tok = blockIdx.x, b = blockIdx.y, c = threadIdx.x;
  const int n = idx[b * NTOK + tok];
  X[(((size_t)(0 * NB + b) * NTOK) + tok) * CCH + c] = fir[((size_t)b * CCH + c) * NPIX + n];
  X[(((size_t)(1 * NB + b) * NTOK) + tok) * CCH + c] = fvis[((size_t)b * CCH + c) * NPIX + n];
  if (c == 0) selw[b * NTOK + tok] = wmap[(size_t)b * NPIX + n];
}

// ---------------------------------------------------------------------------
// LayerNorm(128) -> f16.  One wave per token.
// ---------------------------------------------------------------------------
__global__ void __launch_bounds__(256) ln_kernel(
    const float* __restrict__ X, _Float16* __restrict__ out,
    const float* __restrict__ g_ir, const float* __restrict__ b_ir,
    const float* __restrict__ g_vis, const float* __restrict__ b_vis) {
  const int wave = threadIdx.x >> 5, lane = threadIdx.x & 31;
  const int tok = blockIdx.x * 8 + wave;
  const int sb = blockIdx.y;
  const float4 vv = *(const float4*)(X + (((size_t)sb * NTOK) + tok) * CCH + lane * 4);
  float s = vv.x + vv.y + vv.z + vv.w;
#pragma unroll
  for (int m = 16; m >= 1; m >>= 1) s += __shfl_xor(s, m, 32);
  const float mu = s * (1.0f / 128.0f);
  const float dx = vv.x - mu, dy = vv.y - mu, dz = vv.z - mu, dw = vv.w - mu;
  float q = dx * dx + dy * dy + dz * dz + dw * dw;
#pragma unroll
  for (int m = 16; m >= 1; m >>= 1) q += __shfl_xor(q, m, 32);
  const float inv = rsqrtf(q * (1.0f / 128.0f) + EPSV);
  const float* g = (sb >> 1) ? g_vis : g_ir;
  const float* b = (sb >> 1) ? b_vis : b_ir;
  _Float16* o = out + (((size_t)sb * NTOK) + tok) * CCH + lane * 4;
  const int c = lane * 4;
  o[0] = (_Float16)(dx * inv * g[c + 0] + b[c + 0]);
  o[1] = (_Float16)(dy * inv * g[c + 1] + b[c + 1]);
  o[2] = (_Float16)(dz * inv * g[c + 2] + b[c + 2]);
  o[3] = (_Float16)(dw * inv * g[c + 3] + b[c + 3]);
}

// ---------------------------------------------------------------------------
// Generic token GEMM: C[4096,N] = A16[4096,KA] * Bt16[KA,N] + bias, WMMA core.
// Wave = 16 rows; block 256 = 128 rows; grid.x = 32, grid.y = sb (4).
// ---------------------------------------------------------------------------
struct QkvEpi {
  _Float16 *q, *kt, *v;
  __device__ void operator()(int sb, int m, int n, float val) const {
    if (n < 128) {
      const int h = n >> 5, d = n & 31;
      q[(((size_t)(sb * 4 + h)) * NTOK + m) * 32 + d] = (_Float16)val;
    } else if (n < 256) {
      const int h = (n - 128) >> 5, d = (n - 128) & 31;
      kt[(((size_t)(sb * 4 + h)) * 32 + d) * NTOK + m] = (_Float16)val;  // K stored transposed
    } else {
      const int h = (n - 256) >> 5, d = (n - 256) & 31;
      v[(((size_t)(sb * 4 + h)) * NTOK + m) * 32 + d] = (_Float16)val;
    }
  }
};
struct AddEpi {  // residual accumulate into f32 stream
  float* X;
  __device__ void operator()(int sb, int m, int n, float val) const {
    X[(((size_t)sb * NTOK) + m) * CCH + n] += val;
  }
};
struct GeluEpi {  // exact GELU -> f16 [4096][512]
  _Float16* F;
  __device__ void operator()(int sb, int m, int n, float val) const {
    const float g = 0.5f * val * (1.0f + erff(val * 0.70710678118654752f));
    F[(((size_t)sb * NTOK) + m) * 512 + n] = (_Float16)g;
  }
};

template <int KA, typename Epi>
__global__ void __launch_bounds__(256) gemm_tok_kernel(
    const _Float16* __restrict__ A, const _Float16* __restrict__ Bt,
    const float* __restrict__ bias_ir, const float* __restrict__ bias_vis,
    int N, Epi epi) {
  const int sb = blockIdx.y;
  const int mixer = sb >> 1;
  const int wave = threadIdx.x >> 5, lane = threadIdx.x & 31;
  const int mbase = blockIdx.x * 128 + wave * 16;
  const _Float16* Aw = A + (((size_t)sb * NTOK) + mbase) * KA;
  const _Float16* Bm = Bt + (size_t)mixer * KA * N;
  const float* bias = mixer ? bias_vis : bias_ir;
  __builtin_prefetch(Bm, 0, 1);  // global_prefetch_b8: pull weight tile toward cache

  v16h afr[4];
  if constexpr (KA == 128) {
#pragma unroll
    for (int k = 0; k < 4; ++k) afr[k] = load_a_frag(Aw + 32 * k, KA);
  }

  for (int nt = 0; nt < N / 16; ++nt) {
    v8f acc = {0, 0, 0, 0, 0, 0, 0, 0};
    if constexpr (KA == 128) {
#pragma unroll
      for (int k = 0; k < 4; ++k) {
        v16h b = load_b_frag(Bm + (size_t)(k * 32) * N + nt * 16, N);
        acc = WMMA_F32_F16(afr[k], b, acc);
      }
    } else {
      for (int kc = 0; kc < KA; kc += 32) {
        v16h a = load_a_frag(Aw + kc, KA);
        v16h b = load_b_frag(Bm + (size_t)kc * N + nt * 16, N);
        acc = WMMA_F32_F16(a, b, acc);
      }
    }
    const int n = nt * 16 + (lane & 15);
    const float bv = bias[n];
    const int mrow = mbase + ((lane >> 4) << 3);
#pragma unroll
    for (int r = 0; r < 8; ++r) epi(sb, mrow + r, n, acc[r] + bv);
  }
}

// ---------------------------------------------------------------------------
// Flash attention. Block = 4 waves sharing one (sb, head); per 32-key chunk
// K/V tiles are staged once into LDS (async, ASYNCcnt double-buffered when
// available), cutting K/V global traffic 4x and overlapping fetch with the
// softmax VALU work. P relayout D->A goes through LDS (s_wait_dscnt fence;
// DS ops are in-order per wave).
// ---------------------------------------------------------------------------
static __device__ __forceinline__ void stage_b128(const _Float16* src, _Float16* lds_dst) {
#if HAVE_ASYNC_LDS
  __builtin_amdgcn_global_load_async_to_lds_b128(
      (__attribute__((address_space(1))) v4i*)(void*)src,
      (__attribute__((address_space(3))) v4i*)(void*)lds_dst, 0, 0);
#else
  *(uint4*)lds_dst = *(const uint4*)src;
#endif
}
static __device__ __forceinline__ void wait_stage(bool more_inflight) {
#if HAVE_ASYNC_LDS
  if (more_inflight) __builtin_amdgcn_s_wait_asynccnt(2);
  else               __builtin_amdgcn_s_wait_asynccnt(0);
#else
  (void)more_inflight;
#endif
}

__global__ void __launch_bounds__(128) attn_kernel(
    const _Float16* __restrict__ q, const _Float16* __restrict__ kt,
    const _Float16* __restrict__ v, _Float16* __restrict__ o) {
  extern __shared__ char smem_raw2[];
  _Float16* Ks = (_Float16*)smem_raw2;        // [2][32][40]
  _Float16* Vs = Ks + 2 * 32 * 40;            // [2][32][40]
  _Float16* Pt = Vs + 2 * 32 * 40;            // [4][16][40]
  _Float16* P  = Pt + (size_t)(threadIdx.x >> 5) * (16 * 40);

  const int lane = threadIdx.x & 31;
  const int qb = blockIdx.x * 4 + (threadIdx.x >> 5);
  const int head = blockIdx.y;
  const int sb = blockIdx.z;
  const int hb = sb * 4 + head;
  const _Float16* Q  = q  + (((size_t)hb * NTOK) + qb * 16) * 32;
  const _Float16* KT = kt + ((size_t)hb * 32) * NTOK;
  const _Float16* V  = v  + ((size_t)hb * NTOK) * 32;

  // staging assignment: 128 threads cover 32 rows x 32 cols in 8-half pieces
  const int srow = threadIdx.x >> 2;
  const int scg  = (threadIdx.x & 3) << 3;
  auto issue = [&](int buf, int kc) {
    stage_b128(KT + (size_t)srow * NTOK + kc + scg, Ks + buf * 1280 + srow * 40 + scg);
    stage_b128(V + (size_t)(kc + srow) * 32 + scg, Vs + buf * 1280 + srow * 40 + scg);
  };
  issue(0, 0);

  const v16h aq = load_a_frag(Q, 32);
  const v8f vz = {0, 0, 0, 0, 0, 0, 0, 0};
  v8f o0 = vz, o1 = vz;
  float mr[8], lr[8];
#pragma unroll
  for (int r = 0; r < 8; ++r) { mr[r] = -1e30f; lr[r] = 0.0f; }
  const float scale = 0.17677669529663687f;  // 1/sqrt(32)
  const int prow = (lane >> 4) << 3;
  const int col = lane & 15;

  for (int i = 0; i < NTOK / 32; ++i) {
    const int buf = i & 1;
    const bool more = (i + 1) < (NTOK / 32);
    if (more) issue(buf ^ 1, (i + 1) * 32);  // buf^1 free since end-of-(i-1) barrier
    wait_stage(more);                        // chunk i landed (in-order completion)
    __syncthreads();                         // all waves' portions visible
    const _Float16* Ktile = Ks + buf * 1280;
    const _Float16* Vtile = Vs + buf * 1280;

    v16h bk0 = load_b_frag(Ktile, 40);
    v16h bk1 = load_b_frag(Ktile + 16, 40);
    v8f s0 = WMMA_F32_F16(aq, bk0, vz);
    v8f s1 = WMMA_F32_F16(aq, bk1, vz);
#pragma unroll
    for (int r = 0; r < 8; ++r) {
      const float a0 = s0[r] * scale, a1 = s1[r] * scale;
      float mx = fmaxf(a0, a1);
#pragma unroll
      for (int msk = 8; msk >= 1; msk >>= 1) mx = fmaxf(mx, __shfl_xor(mx, msk, 16));
      const float mn = fmaxf(mr[r], mx);
      const float corr = __expf(mr[r] - mn);
      const float p0 = __expf(a0 - mn), p1 = __expf(a1 - mn);
      float ps = p0 + p1;
#pragma unroll
      for (int msk = 8; msk >= 1; msk >>= 1) ps += __shfl_xor(ps, msk, 16);
      lr[r] = lr[r] * corr + ps;
      mr[r] = mn;
      o0[r] *= corr; o1[r] *= corr;
      P[(prow + r) * 40 + col]      = (_Float16)p0;
      P[(prow + r) * 40 + col + 16] = (_Float16)p1;
    }
    asm volatile("s_wait_dscnt 0x0" ::: "memory");  // P stores visible to this wave's DS reads
    v16h ap  = load_a_frag(P, 40);
    v16h bv0 = load_b_frag(Vtile, 40);
    v16h bv1 = load_b_frag(Vtile + 16, 40);
    o0 = WMMA_F32_F16(ap, bv0, o0);
    o1 = WMMA_F32_F16(ap, bv1, o1);
    __syncthreads();                         // consumers done before buf overwrite
  }

#pragma unroll
  for (int r = 0; r < 8; ++r) {
    const float inv = 1.0f / lr[r];
    const int tok = qb * 16 + prow + r;
    _Float16* dst = o + (((size_t)sb * NTOK) + tok) * CCH + head * 32;
    dst[col]      = (_Float16)(o0[r] * inv);
    dst[16 + col] = (_Float16)(o1[r] * inv);
  }
}

// ---------------------------------------------------------------------------
// Final blend + scatter: out[b][c][idx] = ir*w + vis*(1-w)
// ---------------------------------------------------------------------------
__global__ void scatter_kernel(const float* __restrict__ X, const float* __restrict__ selw,
                               const int* __restrict__ idx, float* __restrict__ out) {
  const int tok = blockIdx.x, b = blockIdx.y, c = threadIdx.x;
  const int n = idx[b * NTOK + tok];
  const float wv = selw[b * NTOK + tok];
  const float ir = X[(((size_t)(0 * NB + b) * NTOK) + tok) * CCH + c];
  const float vs = X[(((size_t)(1 * NB + b) * NTOK) + tok) * CCH + c];
  out[((size_t)b * CCH + c) * NPIX + n] = ir * wv + vs * (1.0f - wv);
}

// ---------------------------------------------------------------------------
// Host launcher
// ---------------------------------------------------------------------------
extern "C" void kernel_launch(void* const* d_in, const int* in_sizes, int n_in,
                              void* d_out, int out_size, void* d_ws, size_t ws_size,
                              hipStream_t stream) {
  (void)in_sizes; (void)n_in; (void)ws_size;
  // Sorted-key pytree flattening of setup_inputs():
  const float* f_ir  = (const float*)d_in[0];
  const float* f_vis = (const float*)d_in[1];
  const float* a_b3  = (const float*)d_in[2];
  const float* bn1_b = (const float*)d_in[3];
  const float* bn1_g = (const float*)d_in[4];
  const float* bn1_m = (const float*)d_in[5];
  const float* bn1_v = (const float*)d_in[6];
  const float* bn2_b = (const float*)d_in[7];
  const float* bn2_g = (const float*)d_in[8];
  const float* bn2_m = (const float*)d_in[9];
  const float* bn2_v = (const float*)d_in[10];
  const float* a_w1  = (const float*)d_in[11];
  const float* a_w3  = (const float*)d_in[12];
  const float* a_wdw = (const float*)d_in[13];
  // 14..17: hypernet (dead code: avg_k always clamps to 4096 for these shapes)
  const int MIX0 = 18, MIX1 = 30;
  // per-mixer sorted offsets: bf1=0 bf2=1 bo=2 bqkv=3 ln1_b=4 ln1_g=5 ln2_b=6 ln2_g=7 wf1=8 wf2=9 wo=10 wqkv=11
  auto mx = [&](int base, int off) { return (const float*)d_in[base + off]; };

  float* out = (float*)d_out;

  // ---- workspace layout (256B aligned) ----
  char* ws = (char*)d_ws;
  size_t off = 0;
  auto alloc = [&](size_t bytes) {
    void* p = ws + off;
    off = (off + bytes + 255) & ~(size_t)255;
    return p;
  };
  float*    x1     = (float*)alloc((size_t)NB * C2 * NPIX * 4);        // 33.6 MB
  float*    wmap   = (float*)alloc((size_t)NB * NPIX * 4);
  float*    sal    = (float*)alloc((size_t)NB * NPIX * 4);
  int*      hist   = (int*)alloc((size_t)NB * 4096 * 4);
  int*      ctrs   = (int*)alloc(64);
  int*      tinfo  = (int*)alloc(64);
  int*      idx    = (int*)alloc((size_t)NB * NTOK * 4);
  float*    selw   = (float*)alloc((size_t)NB * NTOK * 4);
  float*    X      = (float*)alloc((size_t)4 * NTOK * CCH * 4);        // 8.4 MB (residual)
  _Float16* LN16   = (_Float16*)alloc((size_t)4 * NTOK * CCH * 2);
  _Float16* F16    = (_Float16*)alloc((size_t)4 * NTOK * 512 * 2);     // 16.8 MB
  _Float16* O16    = (_Float16*)alloc((size_t)4 * NTOK * CCH * 2);
  _Float16* q16    = (_Float16*)alloc((size_t)16 * NTOK * 32 * 2);
  _Float16* kt16   = (_Float16*)alloc((size_t)16 * 32 * NTOK * 2);
  _Float16* v16    = (_Float16*)alloc((size_t)16 * NTOK * 32 * 2);
  _Float16* w1t    = (_Float16*)alloc((size_t)256 * 64 * 2);
  _Float16* wqkvt  = (_Float16*)alloc((size_t)2 * 128 * 384 * 2);
  _Float16* wot    = (_Float16*)alloc((size_t)2 * 128 * 128 * 2);
  _Float16* wf1t   = (_Float16*)alloc((size_t)2 * 128 * 512 * 2);
  _Float16* wf2t   = (_Float16*)alloc((size_t)2 * 512 * 128 * 2);

  // ---- init + weight prep ----
  zero_kernel<<<32, 256, 0, stream>>>(hist, ctrs, out + (out_size - 1));
  PrepJobs J;
  J.src[0] = a_w1; J.dst[0] = w1t; J.rows[0] = 64; J.cols[0] = 256;
  for (int m = 0; m < 2; ++m) {
    const int base = m ? MIX1 : MIX0;
    const int o = 1 + m * 4;
    J.src[o + 0] = mx(base, 11); J.dst[o + 0] = wqkvt + (size_t)m * 128 * 384; J.rows[o + 0] = 384; J.cols[o + 0] = 128;
    J.src[o + 1] = mx(base, 10); J.dst[o + 1] = wot   + (size_t)m * 128 * 128; J.rows[o + 1] = 128; J.cols[o + 1] = 128;
    J.src[o + 2] = mx(base, 8);  J.dst[o + 2] = wf1t  + (size_t)m * 128 * 512; J.rows[o + 2] = 512; J.cols[o + 2] = 128;
    J.src[o + 3] = mx(base, 9);  J.dst[o + 3] = wf2t  + (size_t)m * 512 * 128; J.rows[o + 3] = 128; J.cols[o + 3] = 512;
  }
  prep_kernel<<<dim3(64, 9), 256, 0, stream>>>(J);

  // ---- dense base: out = f_ir + f_vis ----
  basesum_kernel<<<16384, 256, 0, stream>>>(f_ir, f_vis, out);

  // ---- agent network -> saliency map ----
  agent1_kernel<<<dim3(1024, NB), 128, 64 * 264 * 2, stream>>>(
      f_ir, f_vis, w1t, bn1_b, bn1_g, bn1_m, bn1_v, x1);
  agent2_kernel<<<dim3(16, 16, NB), 256, 0, stream>>>(
      x1, a_wdw, bn2_b, bn2_g, bn2_m, bn2_v, a_w3, a_b3, wmap, sal);

  // ---- exact top-K select + gather ----
  hist_kernel<<<dim3(256, NB), 256, 0, stream>>>(sal, hist);
  thresh_kernel<<<NB, 32, 0, stream>>>(hist, tinfo);
  compact_kernel<<<dim3(256, NB), 256, 0, stream>>>(sal, tinfo, ctrs, idx);
  gather_kernel<<<dim3(NTOK, NB), 128, 0, stream>>>(f_ir, f_vis, wmap, idx, X, selw);

  // ---- mixers (sb = stream*2 + b; both streams in one grid) ----
  ln_kernel<<<dim3(512, 4), 256, 0, stream>>>(X, LN16, mx(MIX0, 5), mx(MIX0, 4), mx(MIX1, 5), mx(MIX1, 4));
  gemm_tok_kernel<128, QkvEpi><<<dim3(32, 4), 256, 0, stream>>>(
      LN16, wqkvt, mx(MIX0, 3), mx(MIX1, 3), 384, QkvEpi{q16, kt16, v16});
  attn_kernel<<<dim3(64, 4, 4), 128, (2 * 1280 + 2 * 1280 + 4 * 640) * 2, stream>>>(
      q16, kt16, v16, O16);
  gemm_tok_kernel<128, AddEpi><<<dim3(32, 4), 256, 0, stream>>>(
      O16, wot, mx(MIX0, 2), mx(MIX1, 2), 128, AddEpi{X});
  ln_kernel<<<dim3(512, 4), 256, 0, stream>>>(X, LN16, mx(MIX0, 7), mx(MIX0, 6), mx(MIX1, 7), mx(MIX1, 6));
  gemm_tok_kernel<128, GeluEpi><<<dim3(32, 4), 256, 0, stream>>>(
      LN16, wf1t, mx(MIX0, 0), mx(MIX1, 0), 512, GeluEpi{F16});
  gemm_tok_kernel<512, AddEpi><<<dim3(32, 4), 256, 0, stream>>>(
      F16, wf2t, mx(MIX0, 1), mx(MIX1, 1), 128, AddEpi{X});

  // ---- blend + scatter ----
  scatter_kernel<<<dim3(NTOK, NB), 128, 0, stream>>>(X, selw, idx, out);
}
